// GlmMoeDsaAttention_71536975282956
// MI455X (gfx1250) — matrix operand order repaired and checked by reading, not compiled, not run on hardware
//
#include <hip/hip_runtime.h>
#include <cmath>

// ---------------------------------------------------------------------------
// GLM MoE DSA attention for gfx1250 (MI455X).
// Matmuls: v_wmma_f32_16x16x32_bf16 (f32 accumulate).
// GEMM A tiles: TDM tensor_load_to_lds (TENSORcnt), double buffered.
// Indexer/flash tiles: GLOBAL_LOAD_ASYNC_TO_LDS_B128 (ASYNCcnt).
// ---------------------------------------------------------------------------

typedef __attribute__((ext_vector_type(16))) __bf16 v16bf;
typedef __attribute__((ext_vector_type(8)))  float  v8f;
typedef __attribute__((ext_vector_type(4)))  unsigned vu4;
typedef __attribute__((ext_vector_type(8)))  int      vi8;
typedef __attribute__((ext_vector_type(4)))  int      vi4;

union FragBF { v16bf v; unsigned u[8]; uint4 q[2]; };

#define DEV __device__ __forceinline__
#define NEGV (-1.0e9f)

DEV unsigned short f2bf(float f) {           // native cvt (RNE in hardware)
  union { __bf16 h; unsigned short u; } t;
  t.h = (__bf16)f;
  return t.u;
}
DEV unsigned pack2bf(float a, float b) {
  union { __bf16 h[2]; unsigned u; } t;
  t.h[0] = (__bf16)a; t.h[1] = (__bf16)b;
  return t.u;
}
DEV v8f wmma_bf16(const FragBF& a, const FragBF& b, v8f c) {
  return __builtin_amdgcn_wmma_f32_16x16x32_bf16(false, a.v, false, b.v,
                                                 (short)0, c, false, false);
}
// Generic LDS pointer -> LDS byte offset (low 32 bits of aperture address).
DEV unsigned ldsoff(const void* p) { return (unsigned)(size_t)p; }
// Async copy 16 bytes global -> LDS (per lane). Tracked by ASYNCcnt.
DEV void async_b128(unsigned lds, const void* gaddr) {
  asm volatile("global_load_async_to_lds_b128 %0, %1, off"
               :: "v"(lds), "v"(gaddr) : "memory");
}
DEV void wait_async0() {
  asm volatile("s_wait_asynccnt 0" ::: "memory");
}

// TDM: DMA a [rows=128] x [cols=32] bf16 tile (row stride lda elements) from
// global to LDS via one Tensor DMA descriptor (D#), per CDNA5 ISA.
DEV void tdm_load_a_tile(unsigned lds_addr, const void* gptr,
                         unsigned lda_elems) {
  unsigned long long ga = (unsigned long long)(size_t)gptr;
  vu4 g0;
  g0[0] = 1u;                                    // count=1, user load
  g0[1] = lds_addr;                              // lds_addr [63:32]
  g0[2] = (unsigned)ga;                          // global_addr low
  g0[3] = (unsigned)((ga >> 32) & 0x1FFFFFFull)  // global_addr[56:32]
          | (2u << 30);                          // type=2 ("image")
  vi8 g1;
  g1[0] = (int)(1u << 16);        // workgroup_mask=0, data_size=1 (2 bytes)
  g1[1] = (int)(32u << 16);       // tensor_dim0=32 (bits 79:48, low half)
  g1[2] = (int)(128u << 16);      // tensor_dim0 hi=0 | tensor_dim1=128 lo
  g1[3] = (int)(32u << 16);       // tensor_dim1 hi=0 | tile_dim0=32
  g1[4] = (int)128u;              // tile_dim1=128, tile_dim2=0
  g1[5] = (int)lda_elems;         // tensor_dim0_stride[31:0]
  g1[6] = 0;                      // stride hi / tensor_dim1_stride
  g1[7] = 0;
  vi4 zz = {};
#if __clang_major__ >= 23
  vi8 z8 = {};
  __builtin_amdgcn_tensor_load_to_lds(g0, g1, zz, zz, z8, 0);
#else
  __builtin_amdgcn_tensor_load_to_lds(g0, g1, zz, zz, 0);
#endif
}

// ---------------------------------------------------------------------------
// f32 -> bf16 conversion kernels (pairwise packed).
// ---------------------------------------------------------------------------
__global__ __launch_bounds__(256) void cvt_bf16_k(
    const float* __restrict__ in, unsigned short* __restrict__ out,
    int npairs) {
  int i = blockIdx.x * 256 + threadIdx.x;
  if (i < npairs) {
    float2 v = *(const float2*)(in + 2 * (size_t)i);
    *(unsigned*)(out + 2 * (size_t)i) = pack2bf(v.x, v.y);
  }
}
__global__ __launch_bounds__(256) void cvt_rows_bf16_k(
    const float* __restrict__ in, int stride, int offset,
    unsigned short* __restrict__ out, int ncols) {
  const float* row = in + (size_t)blockIdx.x * stride + offset;
  unsigned short* orow = out + (size_t)blockIdx.x * ncols;
  for (int c = threadIdx.x * 2; c < ncols; c += blockDim.x * 2) {
    float2 v = *(const float2*)(row + c);
    *(unsigned*)(orow + c) = pack2bf(v.x, v.y);
  }
}

// ---------------------------------------------------------------------------
// GEMM: C[MxN] = A_bf16[MxK] * B_f32[KxN]. 128x128 block, 8 waves (2x4 wmma
// tiles each). A tile: TDM DMA; B tile: convert + transpose. Double buffered:
// next tile's DMA/stage overlaps current tile's WMMA, one barrier per k-step.
// ---------------------------------------------------------------------------
__global__ __launch_bounds__(256) void gemm_bf16_k(
    const unsigned short* __restrict__ A, const float* __restrict__ B,
    float* __restrict__ C, int M, int N, int K, int lda, int ldb, int ldc) {
  __shared__ unsigned short sA[2][128 * 32];   // [row][k] bf16 (row-major)
  __shared__ unsigned short sBT[2][128 * 32];  // [n][k] bf16 (B transposed)
  const int tid = threadIdx.x;
  const int wave = tid >> 5, lane = tid & 31;
  const int half = lane >> 4, r = lane & 15;
  const int wm = (wave & 3) * 32, wn = (wave >> 2) * 64;
  const int m_blk = blockIdx.y * 128, n_blk = blockIdx.x * 128;
  const bool fullN = (n_blk + 128 <= N);       // block-uniform
  (void)M;

  auto loadA = [&](int k0, int buf) {          // one D# per tile, wave 0 only
    if (wave == 0)
      tdm_load_a_tile(ldsoff(&sA[buf][0]),
                      A + (size_t)m_blk * lda + k0, (unsigned)lda);
  };
  auto loadB = [&](int k0, int buf) {
    if (k0 + 32 < K)
      __builtin_prefetch(B + (size_t)(k0 + 32) * ldb + n_blk, 0, 1);
    if (fullN) {                               // vectorized path
#pragma unroll
      for (int i = 0; i < 4; ++i) {
        int f = tid + i * 256;
        int krow = f >> 5, n4 = (f & 31) * 4;
        float4 v = *(const float4*)(B + (size_t)(k0 + krow) * ldb + n_blk + n4);
        sBT[buf][(n4 + 0) * 32 + krow] = f2bf(v.x);
        sBT[buf][(n4 + 1) * 32 + krow] = f2bf(v.y);
        sBT[buf][(n4 + 2) * 32 + krow] = f2bf(v.z);
        sBT[buf][(n4 + 3) * 32 + krow] = f2bf(v.w);
      }
    } else {                                   // guarded tail blocks
#pragma unroll
      for (int i = 0; i < 4; ++i) {
        int f = tid + i * 256;
        int krow = f >> 5, n4 = (f & 31) * 4;
        const float* bp = B + (size_t)(k0 + krow) * ldb + n_blk + n4;
#pragma unroll
        for (int e = 0; e < 4; ++e) {
          float bv = (n_blk + n4 + e < N) ? bp[e] : 0.0f;
          sBT[buf][(n4 + e) * 32 + krow] = f2bf(bv);
        }
      }
    }
  };

  v8f z = {};
  v8f acc[2][4];
#pragma unroll
  for (int a = 0; a < 2; ++a)
#pragma unroll
    for (int b = 0; b < 4; ++b) acc[a][b] = z;

  loadA(0, 0);
  loadB(0, 0);
  if (wave == 0) __builtin_amdgcn_s_wait_tensorcnt((short)0);
  __syncthreads();

  for (int k0 = 0; k0 < K; k0 += 32) {
    const int cur = (k0 >> 5) & 1;
    if (k0 + 32 < K) {                         // overlap next tile with WMMA
      loadA(k0 + 32, cur ^ 1);
      loadB(k0 + 32, cur ^ 1);
    }
    FragBF af[2], bf[4];
#pragma unroll
    for (int tm = 0; tm < 2; ++tm) {
      int mrow = wm + tm * 16 + r;
#pragma unroll
      for (int j = 0; j < 8; ++j) {
        int kk = half * 16 + 2 * j;
        af[tm].u[j] = *(const unsigned*)&sA[cur][mrow * 32 + kk];
      }
    }
#pragma unroll
    for (int tn = 0; tn < 4; ++tn) {
      const uint4* p =
          (const uint4*)&sBT[cur][(wn + tn * 16 + r) * 32 + half * 16];
      bf[tn].q[0] = p[0]; bf[tn].q[1] = p[1];
    }
#pragma unroll
    for (int tm = 0; tm < 2; ++tm)
#pragma unroll
      for (int tn = 0; tn < 4; ++tn)
        acc[tm][tn] = wmma_bf16(af[tm], bf[tn], acc[tm][tn]);
    if (wave == 0) __builtin_amdgcn_s_wait_tensorcnt((short)0);
    __syncthreads();
  }
  if (fullN) {
#pragma unroll
    for (int tm = 0; tm < 2; ++tm)
#pragma unroll
      for (int tn = 0; tn < 4; ++tn)
#pragma unroll
        for (int i = 0; i < 8; ++i) {
          int row = m_blk + wm + tm * 16 + half * 8 + i;
          int col = n_blk + wn + tn * 16 + r;
          C[(size_t)row * ldc + col] = acc[tm][tn][i];
        }
  } else {
#pragma unroll
    for (int tm = 0; tm < 2; ++tm)
#pragma unroll
      for (int tn = 0; tn < 4; ++tn)
#pragma unroll
        for (int i = 0; i < 8; ++i) {
          int row = m_blk + wm + tm * 16 + half * 8 + i;
          int col = n_blk + wn + tn * 16 + r;
          if (col < N) C[(size_t)row * ldc + col] = acc[tm][tn][i];
        }
  }
}

// ---------------------------------------------------------------------------
// In-place RMSNorm over first n columns of each row (row stride `stride`).
// ---------------------------------------------------------------------------
__global__ __launch_bounds__(256) void rmsnorm_k(float* __restrict__ x,
                                                 const float* __restrict__ w,
                                                 int n, int stride) {
  __shared__ float sred[8];
  float* row = x + (size_t)blockIdx.x * stride;
  int tid = threadIdx.x;
  float ss = 0.f;
  for (int i = tid; i < n; i += 256) { float v = row[i]; ss += v * v; }
  for (int m = 16; m; m >>= 1) ss += __shfl_xor(ss, m, 32);
  if ((tid & 31) == 0) sred[tid >> 5] = ss;
  __syncthreads();
  float tot = 0.f;
  for (int wv = 0; wv < 8; ++wv) tot += sred[wv];
  float inv = rsqrtf(tot / n + 1e-6f);
  for (int i = tid; i < n; i += 256) row[i] = row[i] * inv * w[i];
}

// In-place LayerNorm, n == 128 == blockDim.x
__global__ __launch_bounds__(128) void layernorm_k(float* __restrict__ x,
                                                   const float* __restrict__ w,
                                                   const float* __restrict__ b,
                                                   int n) {
  __shared__ float sred[4];
  float* row = x + (size_t)blockIdx.x * n;
  int tid = threadIdx.x;
  float v = row[tid];
  float s = v;
  for (int m = 16; m; m >>= 1) s += __shfl_xor(s, m, 32);
  if ((tid & 31) == 0) sred[tid >> 5] = s;
  __syncthreads();
  float mean = (sred[0] + sred[1] + sred[2] + sred[3]) / n;
  float d = v - mean;
  float s2 = d * d;
  for (int m = 16; m; m >>= 1) s2 += __shfl_xor(s2, m, 32);
  __syncthreads();
  if ((tid & 31) == 0) sred[tid >> 5] = s2;
  __syncthreads();
  float var = (sred[0] + sred[1] + sred[2] + sred[3]) / n;
  row[tid] = d * rsqrtf(var + 1e-6f) * w[tid] + b[tid];
}

// In-place RoPE over 64 dims starting at `offset` of each head.
__global__ void rope_k(float* __restrict__ x, const float* __restrict__ cosb,
                       const float* __restrict__ sinb, int nheads,
                       int head_stride, int row_stride, int offset) {
  int s = blockIdx.x, tid = threadIdx.x;
  int total = nheads * 32;
  for (int it = tid; it < total; it += blockDim.x) {
    int hh = it >> 5, d = it & 31;
    float* p = x + (size_t)s * row_stride + hh * head_stride + offset;
    float c1 = cosb[s * 64 + d],      s1 = sinb[s * 64 + d];
    float c2 = cosb[s * 64 + d + 32], s2 = sinb[s * 64 + d + 32];
    float a = p[d], b = p[d + 32];
    p[d]      = a * c1 - b * s1;     // x1*cos - x2*sin
    p[d + 32] = b * c2 + a * s2;     // x2*cos + x1*sin
  }
}

// ---------------------------------------------------------------------------
// Indexer scores: out[s,t] = sum_h w[s,h]/4 * relu(iq[s,h,:].ik[t,:]/sqrt(128))
//                            + (s>=t ? 0 : NEG)
// Both tiles async-DMAed from pre-converted bf16 tensors (no transpose needed:
// B^T tile == ik rows).
// ---------------------------------------------------------------------------
__global__ __launch_bounds__(256) void index_score_k(
    const unsigned short* __restrict__ iq, const unsigned short* __restrict__ ik,
    const float* __restrict__ w_idx, float* __restrict__ out, int S) {
  const int qb = blockIdx.y * 128, kb = blockIdx.x * 128;
  const int tid = threadIdx.x;
  if (kb > qb) {                              // entirely above diagonal
#pragma unroll
    for (int i = 0; i < 64; ++i) {
      int f = tid + i * 256;
      int s = f >> 7, t = f & 127;
      out[(size_t)(qb + s) * S + kb + t] = NEGV;
    }
    return;
  }
  __shared__ unsigned short sA[128 * 128];    // iq head tile [s][d]
  __shared__ unsigned short sBT[128 * 128];   // ik tile [t][d]
  const int wave = tid >> 5, lane = tid & 31;
  const int half = lane >> 4, r = lane & 15;
  const int wm = (wave & 3) * 32, wn = (wave >> 2) * 64;

#pragma unroll
  for (int i = 0; i < 8; ++i) {               // ik tile: async, loaded once
    int f = tid + i * 256;                    // 2048 x 16B chunks
    int t = f >> 4, ch = (f & 15) * 8;
    async_b128(ldsoff(&sBT[t * 128 + ch]), ik + (size_t)(kb + t) * 128 + ch);
  }
  float fa[2][4][8];
#pragma unroll
  for (int a = 0; a < 2; ++a)
#pragma unroll
    for (int b = 0; b < 4; ++b)
#pragma unroll
      for (int i = 0; i < 8; ++i) fa[a][b][i] = 0.f;
  const float iscale = 0.08838834765f;        // 128^-0.5
  v8f z = {};

  for (int h = 0; h < 16; ++h) {
    __syncthreads();
#pragma unroll
    for (int i = 0; i < 8; ++i) {             // iq tile for this head: async
      int f = tid + i * 256;
      int s = f >> 4, ch = (f & 15) * 8;
      async_b128(ldsoff(&sA[s * 128 + ch]),
                 iq + (size_t)(qb + s) * 2048 + h * 128 + ch);
    }
    wait_async0();
    __syncthreads();
    v8f sacc[2][4];
#pragma unroll
    for (int a = 0; a < 2; ++a)
#pragma unroll
      for (int b = 0; b < 4; ++b) sacc[a][b] = z;
#pragma unroll
    for (int ks = 0; ks < 4; ++ks) {
      FragBF af[2], bf[4];
#pragma unroll
      for (int tm = 0; tm < 2; ++tm) {
        int mrow = wm + tm * 16 + r;
#pragma unroll
        for (int j = 0; j < 8; ++j) {
          int kk = ks * 32 + half * 16 + 2 * j;
          af[tm].u[j] = *(const unsigned*)&sA[mrow * 128 + kk];
        }
      }
#pragma unroll
      for (int tn = 0; tn < 4; ++tn) {
        const uint4* p =
            (const uint4*)&sBT[(wn + tn * 16 + r) * 128 + ks * 32 + half * 16];
        bf[tn].q[0] = p[0]; bf[tn].q[1] = p[1];
      }
#pragma unroll
      for (int tm = 0; tm < 2; ++tm)
#pragma unroll
        for (int tn = 0; tn < 4; ++tn)
          sacc[tm][tn] = wmma_bf16(af[tm], bf[tn], sacc[tm][tn]);
    }
    float wv[2][8];
#pragma unroll
    for (int tm = 0; tm < 2; ++tm)
#pragma unroll
      for (int i = 0; i < 8; ++i)
        wv[tm][i] =
            w_idx[(size_t)(qb + wm + tm * 16 + half * 8 + i) * 16 + h] * 0.25f;
#pragma unroll
    for (int tm = 0; tm < 2; ++tm)
#pragma unroll
      for (int tn = 0; tn < 4; ++tn)
#pragma unroll
        for (int i = 0; i < 8; ++i) {
          float sv = sacc[tm][tn][i] * iscale;
          sv = sv > 0.f ? sv : 0.f;           // relu
          fa[tm][tn][i] += sv * wv[tm][i];
        }
  }
#pragma unroll
  for (int tm = 0; tm < 2; ++tm)
#pragma unroll
    for (int tn = 0; tn < 4; ++tn)
#pragma unroll
      for (int i = 0; i < 8; ++i) {
        int s = qb + wm + tm * 16 + half * 8 + i;
        int t = kb + wn + tn * 16 + r;
        out[(size_t)s * S + t] = fa[tm][tn][i] + (s >= t ? 0.f : NEGV);
      }
}

// ---------------------------------------------------------------------------
// Per-row radix-select: thresh[s] = K-th largest of iscore[s,:].
// allowed(t) == iscore[s,t] >= thresh[s]  (equivalent to topk mask here).
// ---------------------------------------------------------------------------
__global__ __launch_bounds__(256) void topk_thresh_k(
    const float* __restrict__ iscore, float* __restrict__ thresh, int S,
    int K) {
  __shared__ unsigned keys[2048];
  __shared__ unsigned hist[256];
  __shared__ unsigned sh_sel, sh_k;
  int s = blockIdx.x, tid = threadIdx.x;
  for (int i = tid; i < S; i += 256) {
    unsigned u = __float_as_uint(iscore[(size_t)s * S + i]);
    keys[i] = (u & 0x80000000u) ? ~u : (u | 0x80000000u);  // order-preserving
  }
  unsigned prefix = 0, k = (unsigned)K;
  for (int p = 3; p >= 0; --p) {
    __syncthreads();
    hist[tid & 255] = 0;
    __syncthreads();
    for (int i = tid; i < S; i += 256) {
      unsigned key = keys[i];
      bool act = (p == 3) || ((key >> ((p + 1) * 8)) == prefix);
      if (act) atomicAdd(&hist[(key >> (p * 8)) & 255u], 1u);
    }
    __syncthreads();
    if (tid == 0) {
      unsigned cum = 0; int b = 255;
      for (; b > 0; --b) { cum += hist[b]; if (cum >= k) break; }
      if (cum < k) cum += hist[0];
      sh_sel = (unsigned)b;
      sh_k = k - (cum - hist[b]);
    }
    __syncthreads();
    prefix = (prefix << 8) | sh_sel;
    k = sh_k;
  }
  if (tid == 0) {
    unsigned u = (prefix & 0x80000000u) ? (prefix & 0x7FFFFFFFu) : ~prefix;
    thresh[s] = __uint_as_float(u);
  }
}

// ---------------------------------------------------------------------------
// Flash attention. Block = (128 queries, 1 head), 8 waves x 16 query rows.
// Q fragments live in registers (loaded directly from bf16 global);
// K tiles async-DMAed, V^T staged with a manual transpose.
// ---------------------------------------------------------------------------
__global__ __launch_bounds__(256) void flash_attn_k(
    const unsigned short* __restrict__ q,       // [S][16*192] bf16 roped
    const unsigned short* __restrict__ kv_exp,  // [S][16*256] bf16
    const unsigned short* __restrict__ kpe,     // [S][64] bf16 roped
    const float* __restrict__ iscore, const float* __restrict__ thresh,
    float* __restrict__ out, int S) {           // out [S][16*128] f32
  __shared__ unsigned short sK[32 * 192];    // kf tile [t][d]
  __shared__ unsigned short sVT[128 * 32];   // V^T [vd][t]
  __shared__ unsigned short sP[128 * 32];    // probs bf16
  const int h = blockIdx.y;
  const int qb = blockIdx.x * 128;
  const int tid = threadIdx.x;
  const int wave = tid >> 5, lane = tid & 31;
  const int half = lane >> 4, r = lane & 15;
  const int wrow = wave * 16;

  FragBF qfr[6];                              // Q fragments, K=192 => 6 steps
  {
    const unsigned short* qrow = q + (size_t)(qb + wrow + r) * 3072 + h * 192;
#pragma unroll
    for (int ks = 0; ks < 6; ++ks)
#pragma unroll
      for (int j = 0; j < 8; ++j) {
        int kk = ks * 32 + half * 16 + 2 * j;
        qfr[ks].u[j] = *(const unsigned*)(qrow + kk);
      }
  }
  float th_i[8], m_i[8], l_i[8];
  v8f z = {};
  v8f oacc[8];
#pragma unroll
  for (int i = 0; i < 8; ++i) {
    th_i[i] = thresh[qb + wrow + half * 8 + i];
    m_i[i] = -INFINITY; l_i[i] = 0.f; oacc[i] = z;
  }
  const float scale = 0.07216878365f;         // 192^-0.5

  for (int kb = 0; kb < qb + 128; kb += 32) {
    __syncthreads();
#pragma unroll
    for (int i = 0; i < 2; ++i) {             // k_nope 32x128 bf16: async
      int f = tid + i * 256;                  // 512 x 16B chunks
      int t = f >> 4, ch = (f & 15) * 8;
      async_b128(ldsoff(&sK[t * 192 + ch]),
                 kv_exp + (size_t)(kb + t) * 4096 + h * 256 + ch);
    }
    {                                         // k_pe 32x64 bf16: async
      int t = tid >> 3, ch = (tid & 7) * 8;   // 256 x 16B chunks
      async_b128(ldsoff(&sK[t * 192 + 128 + ch]),
                 kpe + (size_t)(kb + t) * 64 + ch);
    }
#pragma unroll
    for (int i = 0; i < 8; ++i) {             // V^T 128x32: manual transpose
      int f = tid + i * 256;                  // 2048 pairs
      int t = f >> 6, n2 = (f & 63) * 2;
      unsigned u = *(const unsigned*)(kv_exp + (size_t)(kb + t) * 4096 +
                                      h * 256 + 128 + n2);
      sVT[(n2 + 0) * 32 + t] = (unsigned short)(u & 0xFFFFu);
      sVT[(n2 + 1) * 32 + t] = (unsigned short)(u >> 16);
    }
    wait_async0();
    __syncthreads();

    v8f sacc[2]; sacc[0] = z; sacc[1] = z;    // S = Q K^T
#pragma unroll
    for (int ks = 0; ks < 6; ++ks)
#pragma unroll
      for (int tn = 0; tn < 2; ++tn) {
        FragBF bf;
        const uint4* p =
            (const uint4*)&sK[(tn * 16 + r) * 192 + ks * 32 + half * 16];
        bf.q[0] = p[0]; bf.q[1] = p[1];
        sacc[tn] = wmma_bf16(qfr[ks], bf, sacc[tn]);
      }
#pragma unroll
    for (int i = 0; i < 8; ++i) {             // streaming softmax per row
      int srow = qb + wrow + half * 8 + i;
      float sv[2];
      float mx = -INFINITY;
#pragma unroll
      for (int tn = 0; tn < 2; ++tn) {
        int tcol = kb + tn * 16 + r;
        float v = sacc[tn][i] * scale;
        if (srow < tcol) v += NEGV;                               // causal
        if (iscore[(size_t)srow * S + tcol] < th_i[i]) v += NEGV; // topk mask
        sv[tn] = v;
        mx = fmaxf(mx, v);
      }
#pragma unroll
      for (int m = 8; m; m >>= 1) mx = fmaxf(mx, __shfl_xor(mx, m, 32));
      float mnew = fmaxf(m_i[i], mx);
      float corr = __expf(m_i[i] - mnew);
      float rs = 0.f;
#pragma unroll
      for (int tn = 0; tn < 2; ++tn) {
        float pv = __expf(sv[tn] - mnew);
        rs += pv;
        sP[(wrow + half * 8 + i) * 32 + tn * 16 + r] = f2bf(pv);
      }
#pragma unroll
      for (int m = 8; m; m >>= 1) rs += __shfl_xor(rs, m, 32);
      l_i[i] = l_i[i] * corr + rs;
      m_i[i] = mnew;
#pragma unroll
      for (int tn = 0; tn < 8; ++tn) oacc[tn][i] *= corr;
    }
    __syncthreads();

    FragBF pf;                                // O += P V (K = 32 keys)
#pragma unroll
    for (int j = 0; j < 8; ++j) {
      int kk = half * 16 + 2 * j;
      pf.u[j] = *(const unsigned*)&sP[(wrow + r) * 32 + kk];
    }
#pragma unroll
    for (int tn = 0; tn < 8; ++tn) {
      FragBF vf;
      const uint4* p = (const uint4*)&sVT[(tn * 16 + r) * 32 + half * 16];
      vf.q[0] = p[0]; vf.q[1] = p[1];
      oacc[tn] = wmma_bf16(pf, vf, oacc[tn]);
    }
  }
#pragma unroll
  for (int tn = 0; tn < 8; ++tn)
#pragma unroll
    for (int i = 0; i < 8; ++i) {
      int row = qb + wrow + half * 8 + i;
      int col = h * 128 + tn * 16 + r;
      out[(size_t)row * 2048 + col] = oacc[tn][i] / l_i[i];
    }
}

// ---------------------------------------------------------------------------
extern "C" void kernel_launch(void* const* d_in, const int* in_sizes, int n_in,
                              void* d_out, int out_size, void* d_ws,
                              size_t ws_size, hipStream_t stream) {
  (void)in_sizes; (void)n_in; (void)out_size; (void)ws_size;
  const int S = 2048;
  const float* x      = (const float*)d_in[0];
  const float* cosb   = (const float*)d_in[1];
  const float* sinb   = (const float*)d_in[2];
  const float* q_a_w      = (const float*)d_in[4];
  const float* q_a_norm_w = (const float*)d_in[5];
  const float* q_b_w      = (const float*)d_in[6];
  const float* kv_a_w     = (const float*)d_in[7];
  const float* kv_a_norm_w= (const float*)d_in[8];
  const float* kv_b_w     = (const float*)d_in[9];
  const float* o_w        = (const float*)d_in[10];
  const float* idx_wq_b_w = (const float*)d_in[11];
  const float* idx_wk_w   = (const float*)d_in[12];
  const float* idx_kn_w   = (const float*)d_in[13];
  const float* idx_kn_b   = (const float*)d_in[14];
  const float* idx_wts_w  = (const float*)d_in[15];

  // f32 intermediates
  float* ws      = (float*)d_ws;
  float* q_resid = ws;                         // S*1536
  float* iq      = q_resid + (size_t)S * 1536; // S*2048
  float* ik      = iq      + (size_t)S * 2048; // S*128
  float* widx    = ik      + (size_t)S * 128;  // S*16
  float* isc     = widx    + (size_t)S * 16;   // S*S
  float* thr     = isc     + (size_t)S * S;    // S
  float* qbuf    = thr     + S;                // S*3072
  float* kv      = qbuf    + (size_t)S * 3072; // S*576
  float* kvex    = kv      + (size_t)S * 576;  // S*4096
  float* aout    = kvex    + (size_t)S * 4096; // S*2048
  // bf16 copies
  unsigned short* xb    = (unsigned short*)(aout + (size_t)S * 2048);
  unsigned short* qrb   = xb   + (size_t)S * 2048; // q_resid bf16
  unsigned short* iqb   = qrb  + (size_t)S * 1536;
  unsigned short* ikb   = iqb  + (size_t)S * 2048;
  unsigned short* qbb   = ikb  + (size_t)S * 128;  // q (roped) bf16
  unsigned short* kvcb  = qbb  + (size_t)S * 3072; // kv_c normed bf16 [S][512]
  unsigned short* kpeb  = kvcb + (size_t)S * 512;  // k_pe roped bf16 [S][64]
  unsigned short* kvexb = kpeb + (size_t)S * 64;
  unsigned short* aoutb = kvexb+ (size_t)S * 4096;

  dim3 blk(256);
  cvt_bf16_k<<<(S * 1024 + 255) / 256, blk, 0, stream>>>(x, xb, S * 1024);
  // q_resid = rmsnorm(x @ q_a_w)
  gemm_bf16_k<<<dim3(12, 16), blk, 0, stream>>>(xb, q_a_w, q_resid,
                                                S, 1536, 2048, 2048, 1536, 1536);
  rmsnorm_k<<<S, 256, 0, stream>>>(q_resid, q_a_norm_w, 1536, 1536);
  cvt_bf16_k<<<(S * 768 + 255) / 256, blk, 0, stream>>>(q_resid, qrb, S * 768);
  // indexer query / key / gate
  gemm_bf16_k<<<dim3(16, 16), blk, 0, stream>>>(qrb, idx_wq_b_w, iq,
                                                S, 2048, 1536, 1536, 2048, 2048);
  rope_k<<<S, 256, 0, stream>>>(iq, cosb, sinb, 16, 128, 2048, 0);
  cvt_bf16_k<<<(S * 1024 + 255) / 256, blk, 0, stream>>>(iq, iqb, S * 1024);
  gemm_bf16_k<<<dim3(1, 16), blk, 0, stream>>>(xb, idx_wk_w, ik,
                                               S, 128, 2048, 2048, 128, 128);
  layernorm_k<<<S, 128, 0, stream>>>(ik, idx_kn_w, idx_kn_b, 128);
  rope_k<<<S, 64, 0, stream>>>(ik, cosb, sinb, 1, 0, 128, 0);
  cvt_bf16_k<<<(S * 64 + 255) / 256, blk, 0, stream>>>(ik, ikb, S * 64);
  gemm_bf16_k<<<dim3(1, 16), blk, 0, stream>>>(xb, idx_wts_w, widx,
                                               S, 16, 2048, 2048, 16, 16);
  // index scores + top-k threshold
  index_score_k<<<dim3(16, 16), blk, 0, stream>>>(iqb, ikb, widx, isc, S);
  topk_thresh_k<<<S, 256, 0, stream>>>(isc, thr, S, 1024);
  // main attention inputs
  gemm_bf16_k<<<dim3(24, 16), blk, 0, stream>>>(qrb, q_b_w, qbuf,
                                                S, 3072, 1536, 1536, 3072, 3072);
  rope_k<<<S, 256, 0, stream>>>(qbuf, cosb, sinb, 16, 192, 3072, 128);
  cvt_bf16_k<<<(S * 1536 + 255) / 256, blk, 0, stream>>>(qbuf, qbb, S * 1536);
  gemm_bf16_k<<<dim3(5, 16), blk, 0, stream>>>(xb, kv_a_w, kv,
                                               S, 576, 2048, 2048, 576, 576);
  rmsnorm_k<<<S, 256, 0, stream>>>(kv, kv_a_norm_w, 512, 576);
  rope_k<<<S, 64, 0, stream>>>(kv, cosb, sinb, 1, 0, 576, 512);
  cvt_rows_bf16_k<<<S, 256, 0, stream>>>(kv, 576, 0, kvcb, 512);
  cvt_rows_bf16_k<<<S, 64, 0, stream>>>(kv, 576, 512, kpeb, 64);
  gemm_bf16_k<<<dim3(32, 16), blk, 0, stream>>>(kvcb, kv_b_w, kvex,
                                                S, 4096, 512, 512, 4096, 4096);
  cvt_bf16_k<<<(S * 2048 + 255) / 256, blk, 0, stream>>>(kvex, kvexb, S * 2048);
  // flash attention + output projection
  flash_attn_k<<<dim3(16, 16), blk, 0, stream>>>(qbb, kvexb, kpeb, isc, thr,
                                                 aout, S);
  cvt_bf16_k<<<(S * 1024 + 255) / 256, blk, 0, stream>>>(aout, aoutb, S * 1024);
  gemm_bf16_k<<<dim3(16, 16), blk, 0, stream>>>(aoutb, o_w, (float*)d_out,
                                                S, 2048, 2048, 2048, 2048, 2048);
}
